// Qwen3Attention_86500641341526
// MI455X (gfx1250) — matrix operand order repaired and checked by reading, not compiled, not run on hardware
//
#include <hip/hip_runtime.h>
#include <hip/hip_bf16.h>

// ---------------- constants ----------------
#define TSEQ   4096
#define HID    2048
#define NHEADS 16
#define NKV    8
#define HDIM   128
#define QSIZE  (NHEADS*HDIM)        // 2048
#define KVSIZE (NKV*HDIM)           // 1024
#define QKVW   (QSIZE + 2*KVSIZE)   // 4096

typedef __attribute__((ext_vector_type(16))) __bf16 v16bf;
typedef __attribute__((ext_vector_type(8)))  float  v8f;

// ---------------- helpers ----------------
__device__ __forceinline__ __bf16 to_bf16(float f) {
  unsigned u; __builtin_memcpy(&u, &f, 4);
  u += 0x7FFFu + ((u >> 16) & 1u);          // round-to-nearest-even
  unsigned short s = (unsigned short)(u >> 16);
  __bf16 r; __builtin_memcpy(&r, &s, 2);
  return r;
}
__device__ __forceinline__ float bf16_to_f(__bf16 h) {
  unsigned short s; __builtin_memcpy(&s, &h, 2);
  unsigned u = ((unsigned)s) << 16;
  float f; __builtin_memcpy(&f, &u, 4);
  return f;
}

// A-fragment: 16x32 bf16, row-major source with leading dim ld.
// lane L: m = L&15, g = L>>4; elem i -> K = kc + 16g + i  (contiguous 16)
__device__ __forceinline__ v16bf load_a_frag(const __bf16* __restrict__ src, int ld, int kc) {
  int lane = threadIdx.x & 31;
  int m = lane & 15, g = lane >> 4;
  const __bf16* p = src + (size_t)m * ld + kc + 16 * g;
  v16bf a;
#pragma unroll
  for (int i = 0; i < 8; ++i) a[i] = p[i];
#pragma unroll
  for (int i = 0; i < 8; ++i) a[8 + i] = p[8 + i];
  return a;
}

// B-fragment: 32x16 bf16 from column-major storage bt[n][k] (ld = K).
// lane L: n = L&15, g = L>>4; elem i -> K = kc + 16g + i  (contiguous 16)
__device__ __forceinline__ v16bf load_bt_frag(const __bf16* __restrict__ bt, int ld, int kc) {
  int lane = threadIdx.x & 31;
  int n = lane & 15, g = lane >> 4;
  const __bf16* p = bt + (size_t)n * ld + kc + 16 * g;
  v16bf b;
#pragma unroll
  for (int i = 0; i < 16; ++i) b[i] = p[i];
  return b;
}

// ---------------- conversion kernels ----------------
__global__ void __launch_bounds__(256, 1)
cvt_kernel(const float* __restrict__ src, __bf16* __restrict__ dst, int n) {
  int i = blockIdx.x * blockDim.x + threadIdx.x;
  if (i < n) dst[i] = to_bf16(src[i]);
}

// dst[c][r] = src[r][c]  (src R x C row-major  ->  dst C x R row-major)
__global__ void __launch_bounds__(256, 1)
cvt_t_kernel(const float* __restrict__ src, __bf16* __restrict__ dst, int R, int C) {
  int i = blockIdx.x * blockDim.x + threadIdx.x;
  if (i < R * C) {
    int r = i / C, c = i % C;
    dst[(size_t)c * R + r] = to_bf16(src[i]);
  }
}

// ---------------- generic bf16 GEMM: C[MxN] = A[MxK] * Bt[NxK]^T ----------------
// One wave computes a 16x64 strip (A-fragment reuse x4).
template <bool BF16OUT>
__global__ void __launch_bounds__(256, 1)
gemm_kernel(const __bf16* __restrict__ A, const __bf16* __restrict__ Bt,
            void* __restrict__ Cout, int M, int N, int K) {
  int wid = blockIdx.x * (blockDim.x >> 5) + (threadIdx.x >> 5);
  int tiles_n4 = N >> 6;                 // 64-wide strips
  int mt  = wid / tiles_n4;
  int nt4 = wid % tiles_n4;
  if (mt >= (M >> 4)) return;            // wave-uniform
  int lane = threadIdx.x & 31;
  int n = lane & 15, g = lane >> 4;

  v8f c[4] = {};
  const __bf16* arow = A + (size_t)(mt * 16) * K;
  const __bf16* brow = Bt + (size_t)(nt4 * 64) * K;
#pragma unroll 2
  for (int kc = 0; kc < K; kc += 32) {
    v16bf a = load_a_frag(arow, K, kc);
    if (kc + 64 < K) {                   // prefetch next K-slab (global_prefetch_b8)
      __builtin_prefetch(arow + (size_t)(lane & 15) * K + kc + 64, 0, 1);
      __builtin_prefetch(brow + (size_t)lane * K + kc + 64, 0, 1);
    }
#pragma unroll
    for (int t = 0; t < 4; ++t) {
      v16bf b = load_bt_frag(brow + (size_t)(t * 16) * K, K, kc);
      c[t] = __builtin_amdgcn_wmma_f32_16x16x32_bf16(false, a, false, b, (short)0, c[t], false, false);
    }
  }
#pragma unroll
  for (int t = 0; t < 4; ++t) {
    size_t col = (size_t)nt4 * 64 + t * 16 + n;
#pragma unroll
    for (int j = 0; j < 8; ++j) {
      size_t row = (size_t)mt * 16 + j + 8 * g;
      if (BF16OUT) ((__bf16*)Cout)[row * N + col] = to_bf16(c[t][j]);
      else         ((float*)Cout)[row * N + col]  = c[t][j];
    }
  }
}

// ---------------- RMSNorm + RoPE + layout kernel ----------------
// grid: (TSEQ, 32). slot<16: Q head; slot<24: K head; else V head.
// Qb[h][t][d], Kb[h][t][d] row-major bf16 ; Vt[h][d][t] (transposed) bf16.
__global__ void __launch_bounds__(128, 1)
norm_rope_kernel(const int* __restrict__ positions,
                 const __bf16* __restrict__ qkv,
                 const float* __restrict__ qw,
                 const float* __restrict__ kw,
                 __bf16* __restrict__ Qb,
                 __bf16* __restrict__ Kb,
                 __bf16* __restrict__ Vt) {
  int t = blockIdx.x;
  int slot = blockIdx.y;
  int d = threadIdx.x;                    // 0..127
  const __bf16* row = qkv + (size_t)t * QKVW;

  __shared__ float xs[HDIM];
  __shared__ float red[4];

  if (slot < 24) {                        // Q or K: rmsnorm + rope
    bool isQ = (slot < 16);
    int h = isQ ? slot : (slot - 16);
    int off = isQ ? h * HDIM : QSIZE + h * HDIM;
    const float* w = isQ ? qw : kw;

    float x = bf16_to_f(row[off + d]);
    float ss = x * x;
#pragma unroll
    for (int m = 16; m > 0; m >>= 1) ss += __shfl_xor(ss, m, 32);
    if ((threadIdx.x & 31) == 0) red[threadIdx.x >> 5] = ss;
    __syncthreads();
    float tot = red[0] + red[1] + red[2] + red[3];
    float r = rsqrtf(tot * (1.0f / HDIM) + 1e-6f);
    xs[d] = x * r * w[d];
    __syncthreads();

    float pos = (float)positions[t];
    int dr = d & 63;
    float invf = __powf(10000.0f, -(float)dr * (1.0f / 64.0f));
    float f = pos * invf;
    float cs = __cosf(f), sn = __sinf(f);
    float outv = (d < 64) ? (xs[d] * cs - xs[d + 64] * sn)
                          : (xs[d] * cs + xs[d - 64] * sn);
    __bf16 ob = to_bf16(outv);
    if (isQ) Qb[((size_t)h * TSEQ + t) * HDIM + d] = ob;
    else     Kb[((size_t)h * TSEQ + t) * HDIM + d] = ob;
  } else {                                // V: just scatter transposed
    int h = slot - 24;
    Vt[((size_t)h * HDIM + d) * TSEQ + t] = row[QSIZE + KVSIZE + h * HDIM + d];
  }
}

// ---------------- flash attention: one wave per (head, 16-query tile) ----------------
// 4 waves / block so each wave gets the full VGPR budget (no spills).
__global__ void __launch_bounds__(128, 1)
attn_kernel(const __bf16* __restrict__ Qb,
            const __bf16* __restrict__ Kb,
            const __bf16* __restrict__ Vt,
            __bf16* __restrict__ attnb) {
  int widb = threadIdx.x >> 5;
  int wid = blockIdx.x * (blockDim.x >> 5) + widb;
  int h  = wid >> 8;                      // 256 query tiles per head
  int qt = wid & 255;
  int kvh = h >> 1;                       // GQA: 2 q-heads per kv-head
  int lane = threadIdx.x & 31;
  int n = lane & 15, g = lane >> 4;

  __shared__ __bf16 pbuf[4][16 * 32];     // per-wave probability tile (bf16)
  __bf16* pw = pbuf[widb];

  // Q fragments for this tile: 4 chunks of 32 along HDIM
  const __bf16* qbase = Qb + ((size_t)h * TSEQ + qt * 16) * HDIM;
  v16bf qa[4];
#pragma unroll
  for (int kc = 0; kc < 4; ++kc) qa[kc] = load_a_frag(qbase, HDIM, kc * 32);

  v8f o[8] = {};
  float rmax[8], rsum[8];
#pragma unroll
  for (int j = 0; j < 8; ++j) { rmax[j] = -1e30f; rsum[j] = 0.0f; }

  const float scale = 0.08838834764831845f;   // 1/sqrt(128)
  int qend = qt * 16 + 15;
  int nblocks = qend / 32 + 1;

  for (int b = 0; b < nblocks; ++b) {
    int kb = b * 32;
    // ---- scores S = Q * K^T for 32 keys (2 tiles of 16) ----
    v8f s0 = {}, s1 = {};
    const __bf16* kb0 = Kb + ((size_t)kvh * TSEQ + kb) * HDIM;
    const __bf16* kb1 = kb0 + 16 * HDIM;
#pragma unroll
    for (int kc = 0; kc < 4; ++kc) {
      v16bf bk0 = load_bt_frag(kb0, HDIM, kc * 32);
      v16bf bk1 = load_bt_frag(kb1, HDIM, kc * 32);
      s0 = __builtin_amdgcn_wmma_f32_16x16x32_bf16(false, qa[kc], false, bk0, (short)0, s0, false, false);
      s1 = __builtin_amdgcn_wmma_f32_16x16x32_bf16(false, qa[kc], false, bk1, (short)0, s1, false, false);
    }
    // ---- online softmax; exponentials written straight into the LDS P-tile ----
    __builtin_amdgcn_wave_barrier();      // previous P*V fragment reads are done
    float cf[8];
#pragma unroll
    for (int j = 0; j < 8; ++j) {
      int rowq = qt * 16 + j + 8 * g;
      float v0 = (kb + n      <= rowq) ? s0[j] * scale : -1e30f;
      float v1 = (kb + 16 + n <= rowq) ? s1[j] * scale : -1e30f;
      float bm = fmaxf(v0, v1);
      bm = fmaxf(bm, __shfl_xor(bm, 1, 32));
      bm = fmaxf(bm, __shfl_xor(bm, 2, 32));
      bm = fmaxf(bm, __shfl_xor(bm, 4, 32));
      bm = fmaxf(bm, __shfl_xor(bm, 8, 32));
      float nm = fmaxf(rmax[j], bm);
      cf[j] = __expf(rmax[j] - nm);
      rmax[j] = nm;
      float e0 = __expf(v0 - nm);
      float e1 = __expf(v1 - nm);
      pw[(j + 8 * g) * 32 + n]      = to_bf16(e0);
      pw[(j + 8 * g) * 32 + 16 + n] = to_bf16(e1);
      float bs = e0 + e1;
      bs += __shfl_xor(bs, 1, 32);
      bs += __shfl_xor(bs, 2, 32);
      bs += __shfl_xor(bs, 4, 32);
      bs += __shfl_xor(bs, 8, 32);
      rsum[j] = rsum[j] * cf[j] + bs;
    }
    // rescale accumulator rows by correction factor
#pragma unroll
    for (int t2 = 0; t2 < 8; ++t2)
#pragma unroll
      for (int j = 0; j < 8; ++j) o[t2][j] *= cf[j];

    asm volatile("s_wait_dscnt 0" ::: "memory");
    __builtin_amdgcn_wave_barrier();
    // ---- read P back in A-fragment layout ----
    v16bf pa;
    {
      const __bf16* pp = pw + n * 32 + 16 * g;   // m = lane&15
#pragma unroll
      for (int i = 0; i < 8; ++i) { pa[i] = pp[i]; pa[8 + i] = pp[8 + i]; }
    }
    // ---- O += P * V  (V transposed per head: contiguous B-fragments) ----
    const __bf16* vb = Vt + (size_t)kvh * HDIM * TSEQ + kb;
#pragma unroll
    for (int t2 = 0; t2 < 8; ++t2) {
      const __bf16* pv = vb + (size_t)(t2 * 16 + n) * TSEQ + 16 * g;
      v16bf bv;
#pragma unroll
      for (int i = 0; i < 16; ++i) bv[i] = pv[i];
      o[t2] = __builtin_amdgcn_wmma_f32_16x16x32_bf16(false, pa, false, bv, (short)0, o[t2], false, false);
    }
  }
  // ---- normalize and store bf16 [token][h*128 + dim] ----
#pragma unroll
  for (int t2 = 0; t2 < 8; ++t2) {
#pragma unroll
    for (int j = 0; j < 8; ++j) {
      size_t rowq = (size_t)qt * 16 + j + 8 * g;
      attnb[rowq * QSIZE + h * HDIM + t2 * 16 + n] = to_bf16(o[t2][j] / rsum[j]);
    }
  }
}

// ---------------- host launcher ----------------
extern "C" void kernel_launch(void* const* d_in, const int* in_sizes, int n_in,
                              void* d_out, int out_size, void* d_ws, size_t ws_size,
                              hipStream_t stream) {
  const int*   positions = (const int*)d_in[0];
  const float* hidden    = (const float*)d_in[1];
  const float* w_qkv     = (const float*)d_in[2];
  const float* q_norm_w  = (const float*)d_in[3];
  const float* k_norm_w  = (const float*)d_in[4];
  const float* w_o       = (const float*)d_in[5];
  float* out = (float*)d_out;

  char* ws = (char*)d_ws;
  const size_t MB = 1u << 20;
  __bf16* Ah     = (__bf16*)(ws +   0 * MB);   // hidden bf16       16 MB
  __bf16* WqkvT  = (__bf16*)(ws +  16 * MB);   // w_qkv^T bf16      16 MB
  __bf16* WoT    = (__bf16*)(ws +  32 * MB);   // w_o^T bf16         8 MB
  __bf16* qkvb   = (__bf16*)(ws +  40 * MB);   // qkv bf16          32 MB
  __bf16* Qb     = (__bf16*)(ws +  72 * MB);   // Q per-head        16 MB
  __bf16* Kbuf   = (__bf16*)(ws +  88 * MB);   // K per-head         8 MB
  __bf16* Vtb    = (__bf16*)(ws +  96 * MB);   // V^T per-head       8 MB
  __bf16* attnb  = (__bf16*)(ws + 104 * MB);   // attention out     16 MB

  // 1) precision conversion / transposition
  {
    int n = TSEQ * HID;
    cvt_kernel<<<(n + 255) / 256, 256, 0, stream>>>(hidden, Ah, n);
  }
  cvt_t_kernel<<<(HID * QKVW + 255) / 256, 256, 0, stream>>>(w_qkv, WqkvT, HID, QKVW);
  cvt_t_kernel<<<(QSIZE * HID + 255) / 256, 256, 0, stream>>>(w_o, WoT, QSIZE, HID);

  // 2) QKV projection (bf16 WMMA): 4096 x 4096 x 2048
  {
    int waves = (TSEQ / 16) * (QKVW / 64);
    gemm_kernel<true><<<waves / 8, 256, 0, stream>>>(Ah, WqkvT, qkvb, TSEQ, QKVW, HID);
  }

  // 3) RMSNorm + RoPE + per-head layout (V transposed)
  norm_rope_kernel<<<dim3(TSEQ, 32), 128, 0, stream>>>(positions, qkvb, q_norm_w, k_norm_w,
                                                       Qb, Kbuf, Vtb);

  // 4) causal flash attention per (head, 16-row query tile), 4 waves/block
  {
    int waves = NHEADS * (TSEQ / 16);
    attn_kernel<<<waves / 4, 128, 0, stream>>>(Qb, Kbuf, Vtb, attnb);
  }

  // 5) output projection -> fp32 d_out: 4096 x 2048 x 2048
  {
    int waves = (TSEQ / 16) * (HID / 64);
    gemm_kernel<false><<<waves / 8, 256, 0, stream>>>(attnb, WoT, out, TSEQ, HID, HID);
  }
}
